// CQAttention_2920577761511
// MI455X (gfx1250) — matrix operand order repaired and checked by reading, not compiled, not run on hardware
//
#include <hip/hip_runtime.h>
#include <hip/hip_bf16.h>

typedef __bf16 bf16;
typedef __attribute__((ext_vector_type(16))) __bf16 v16bf;
typedef __attribute__((ext_vector_type(8)))  __bf16 v8bf;
typedef __attribute__((ext_vector_type(8)))  float  v8f;

constexpr int B  = 32;
constexpr int H  = 256;
constexpr int Lc = 1024;
constexpr int Lq = 256;
constexpr float NEGV = -1e30f;

// ---------------- WMMA fragment helpers (wave32, 16x16x32 bf16) ----------------
__device__ __forceinline__ v16bf load_a_frag(const bf16* A, int lda, int row0, int k0, int lane) {
    int r  = row0 + (lane & 15);
    int kb = k0 + ((lane >> 4) << 3);
    const bf16* p = A + (size_t)r * lda + kb;
    v8bf lo = *(const v8bf*)(p);
    v8bf hi = *(const v8bf*)(p + 16);
    v16bf o;
#pragma unroll
    for (int i = 0; i < 8; ++i) { o[i] = lo[i]; o[i + 8] = hi[i]; }
    return o;
}
__device__ __forceinline__ v16bf load_b_frag(const bf16* BT, int ldb, int col0, int k0, int lane) {
    int n  = col0 + (lane & 15);
    int kb = k0 + ((lane >> 4) << 3);
    const bf16* p = BT + (size_t)n * ldb + kb;
    return join8(*(const v8bf*)(p), *(const v8bf*)(p + 16));
}
__device__ __forceinline__ v16bf join8(v8bf lo, v8bf hi) {
    v16bf o;
#pragma unroll
    for (int i = 0; i < 8; ++i) { o[i] = lo[i]; o[i + 8] = hi[i]; }
    return o;
}
__device__ __forceinline__ v8bf mul8(const bf16* x, const bf16* y) {
    v8bf a = *(const v8bf*)x, b = *(const v8bf*)y, r;
#pragma unroll
    for (int i = 0; i < 8; ++i) r[i] = (bf16)((float)a[i] * (float)b[i]);
    return r;
}

// ---------------- K0: W_res f32 -> bf16 ----------------
__global__ void conv_w_kernel(const float* __restrict__ W, bf16* __restrict__ Wb, int n) {
    int i = blockIdx.x * blockDim.x + threadIdx.x;
    if (i < n) Wb[i] = (bf16)W[i];
}

// ---------------- K1: transpose/convert (B,H,L) f32 -> (B,L,H) bf16 -----------
__global__ __launch_bounds__(256) void trans_kernel(const float* __restrict__ src,
                                                    const float* __restrict__ w3,
                                                    bf16* __restrict__ dstT,
                                                    bf16* __restrict__ dstTw,
                                                    bf16* __restrict__ dstSame, int L) {
    __shared__ float tile[32][33];
    int t = threadIdx.x, tx = t & 31, ty = t >> 5;
    int LT = L / 32;
    int bid = blockIdx.x;
    int lt = bid % LT, ht = (bid / LT) % (H / 32), b = bid / (LT * (H / 32));
    int h0 = ht * 32, l0 = lt * 32;
    const float* sb = src + (size_t)b * H * L;
#pragma unroll
    for (int j = 0; j < 4; ++j) {
        int h = h0 + ty + 8 * j;
        float v = sb[(size_t)h * L + l0 + tx];
        tile[ty + 8 * j][tx] = v;
        dstSame[(size_t)b * H * L + (size_t)h * L + l0 + tx] = (bf16)v;
    }
    __syncthreads();
#pragma unroll
    for (int j = 0; j < 4; ++j) {
        int l = l0 + ty + 8 * j;
        float v = tile[tx][ty + 8 * j];
        size_t o = ((size_t)b * L + l) * H + h0 + tx;
        dstT[o] = (bf16)v;
        if (dstTw) dstTw[o] = (bf16)(v * w3[h0 + tx]);
    }
}

// ---------------- K2: c1[b,l] = Ct[l]·w1 ; q2[b,m] = Qt[m]·w2 ----------------
__global__ void vec_kernel(const float* __restrict__ C, const float* __restrict__ Q,
                           const float* __restrict__ w, float* __restrict__ c1,
                           float* __restrict__ q2) {
    int i = blockIdx.x * blockDim.x + threadIdx.x;
    if (i < B * Lc) {
        int b = i / Lc, l = i % Lc;
        float s = 0.f;
        for (int h = 0; h < H; ++h) s += C[((size_t)b * H + h) * Lc + l] * w[h];
        c1[i] = s;
    } else if (i < B * Lc + B * Lq) {
        int j = i - B * Lc, b = j / Lq, m = j % Lq;
        float s = 0.f;
        for (int h = 0; h < H; ++h) s += Q[((size_t)b * H + h) * Lq + m] * w[H + h];
        q2[j] = s;
    }
}

// ---------------- K3: S = CWb(Lc,H) x Qtb(Lq,H)^T + c1 + q2 (f32 out) ---------
// Each wave computes a 16(M) x 64(N) strip: 1 A-frag feeds 4 WMMAs per k-step.
__global__ __launch_bounds__(256) void gemm_score(const bf16* __restrict__ CW,
                                                  const bf16* __restrict__ QT,
                                                  const float* __restrict__ c1,
                                                  const float* __restrict__ q2,
                                                  float* __restrict__ S) {
    int lane = threadIdx.x & 31;
    int wave = blockIdx.x * 8 + (threadIdx.x >> 5);
    const int MT = Lc / 16, NB = Lq / 64;
    int nb = wave % NB, mt = (wave / NB) % MT, b = wave / (NB * MT);
    const bf16* A  = CW + (size_t)b * Lc * H;
    const bf16* BT = QT + (size_t)b * Lq * H;
    int m0 = mt * 16, n0 = nb * 64;
    v8f acc0 = {}, acc1 = {}, acc2 = {}, acc3 = {};
#pragma unroll 2
    for (int k0 = 0; k0 < H; k0 += 32) {
        v16bf a  = load_a_frag(A, H, m0, k0, lane);
        v16bf b0 = load_b_frag(BT, H, n0,      k0, lane);
        v16bf b1 = load_b_frag(BT, H, n0 + 16, k0, lane);
        v16bf b2 = load_b_frag(BT, H, n0 + 32, k0, lane);
        v16bf b3 = load_b_frag(BT, H, n0 + 48, k0, lane);
        acc0 = __builtin_amdgcn_wmma_f32_16x16x32_bf16(false, a, false, b0, (short)0, acc0, false, false);
        acc1 = __builtin_amdgcn_wmma_f32_16x16x32_bf16(false, a, false, b1, (short)0, acc1, false, false);
        acc2 = __builtin_amdgcn_wmma_f32_16x16x32_bf16(false, a, false, b2, (short)0, acc2, false, false);
        acc3 = __builtin_amdgcn_wmma_f32_16x16x32_bf16(false, a, false, b3, (short)0, acc3, false, false);
    }
    int hi = lane >> 4;
    float* Sb = S + (size_t)b * Lc * Lq;
    v8f accs[4] = {acc0, acc1, acc2, acc3};
#pragma unroll
    for (int j = 0; j < 4; ++j) {
        int n = n0 + 16 * j + (lane & 15);
        float qv = q2[b * Lq + n];
#pragma unroll
        for (int r = 0; r < 8; ++r) {
            int m = m0 + r + 8 * hi;
            Sb[(size_t)m * Lq + n] = accs[j][r] + c1[b * Lc + m] + qv;
        }
    }
}

// ---------------- K4: row softmax over m (wave per row) -> Srow bf16 ----------
__global__ __launch_bounds__(256) void row_softmax(const float* __restrict__ S,
                                                   const int* __restrict__ qmask,
                                                   bf16* __restrict__ Srow) {
    int lane = threadIdx.x & 31;
    int row = blockIdx.x * 8 + (threadIdx.x >> 5);   // b*Lc + l
    int b = row / Lc;
    const float* r = S + (size_t)row * Lq;
    float v[8], mx = -3.0e38f;
#pragma unroll
    for (int i = 0; i < 8; ++i) {
        int m = lane + 32 * i;
        float s = r[m];
        if (!qmask[b * Lq + m]) s += NEGV;
        v[i] = s; mx = fmaxf(mx, s);
    }
    for (int o = 16; o >= 1; o >>= 1) mx = fmaxf(mx, __shfl_xor(mx, o, 32));
    float e[8], sum = 0.f;
#pragma unroll
    for (int i = 0; i < 8; ++i) { e[i] = __expf(v[i] - mx); sum += e[i]; }
    for (int o = 16; o >= 1; o >>= 1) sum += __shfl_xor(sum, o, 32);
    float inv = 1.0f / sum;
    bf16* orow = Srow + (size_t)row * Lq;
#pragma unroll
    for (int i = 0; i < 8; ++i) orow[lane + 32 * i] = (bf16)(e[i] * inv);
}

// ---------------- K5: col softmax over l + transpose -> ScolT[b,m,l] bf16 -----
__global__ __launch_bounds__(256) void col_softmax_t(const float* __restrict__ S,
                                                     const int* __restrict__ cmask,
                                                     bf16* __restrict__ ScolT) {
    __shared__ float red[8][32];
    __shared__ float colmax[32], colinv[32];
    __shared__ float ltile[32][33];
    int t = threadIdx.x, tx = t & 31, tp = t >> 5;
    int nb = Lq / 32;
    int mb = blockIdx.x % nb, b = blockIdx.x / nb;
    int m0 = mb * 32;
    const float* Sb = S + (size_t)b * Lc * Lq;
    const int* cm = cmask + b * Lc;

    float mx = -3.0e38f;
    for (int i = 0; i < Lc / 8; ++i) {
        int l = tp * (Lc / 8) + i;
        float s = Sb[(size_t)l * Lq + m0 + tx];
        if (!cm[l]) s += NEGV;
        mx = fmaxf(mx, s);
    }
    red[tp][tx] = mx; __syncthreads();
    if (tp == 0) {
        float m2 = red[0][tx];
        for (int j = 1; j < 8; ++j) m2 = fmaxf(m2, red[j][tx]);
        colmax[tx] = m2;
    }
    __syncthreads();
    float cmx = colmax[tx], sm = 0.f;
    for (int i = 0; i < Lc / 8; ++i) {
        int l = tp * (Lc / 8) + i;
        float s = Sb[(size_t)l * Lq + m0 + tx];
        if (!cm[l]) s += NEGV;
        sm += __expf(s - cmx);
    }
    red[tp][tx] = sm; __syncthreads();
    if (tp == 0) {
        float s2 = 0.f;
        for (int j = 0; j < 8; ++j) s2 += red[j][tx];
        colinv[tx] = 1.0f / s2;
    }
    __syncthreads();
    float cin = colinv[tx];
    for (int l0 = 0; l0 < Lc; l0 += 32) {
#pragma unroll
        for (int j = 0; j < 4; ++j) {
            int l = l0 + tp + 8 * j;
            float s = Sb[(size_t)l * Lq + m0 + tx];
            if (!cm[l]) s += NEGV;
            ltile[tp + 8 * j][tx] = __expf(s - cmx) * cin;
        }
        __syncthreads();
#pragma unroll
        for (int j = 0; j < 4; ++j) {
            int ml = tp + 8 * j;
            ScolT[((size_t)b * Lq + m0 + ml) * Lc + l0 + tx] = (bf16)ltile[tx][ml];
        }
        __syncthreads();
    }
}

// ------- generic bf16 GEMM: D(MxN) = A(MxK) x BT(NxK)^T, 16x64 per wave -------
__global__ __launch_bounds__(256) void gemm_bf(const bf16* __restrict__ A,
                                               const bf16* __restrict__ BT,
                                               bf16* __restrict__ D,
                                               int M, int N, int K,
                                               long sA, long sB, long sD) {
    int lane = threadIdx.x & 31;
    int wave = blockIdx.x * 8 + (threadIdx.x >> 5);
    int MT = M / 16, NB = N / 64;
    int nb = wave % NB, mt = (wave / NB) % MT, b = wave / (NB * MT);
    const bf16* Ab = A + (size_t)b * sA;
    const bf16* Bb = BT + (size_t)b * sB;
    bf16* Db = D + (size_t)b * sD;
    int m0 = mt * 16, n0 = nb * 64;
    v8f acc0 = {}, acc1 = {}, acc2 = {}, acc3 = {};
#pragma unroll 2
    for (int k0 = 0; k0 < K; k0 += 32) {
        v16bf a  = load_a_frag(Ab, K, m0, k0, lane);
        v16bf b0 = load_b_frag(Bb, K, n0,      k0, lane);
        v16bf b1 = load_b_frag(Bb, K, n0 + 16, k0, lane);
        v16bf b2 = load_b_frag(Bb, K, n0 + 32, k0, lane);
        v16bf b3 = load_b_frag(Bb, K, n0 + 48, k0, lane);
        acc0 = __builtin_amdgcn_wmma_f32_16x16x32_bf16(false, a, false, b0, (short)0, acc0, false, false);
        acc1 = __builtin_amdgcn_wmma_f32_16x16x32_bf16(false, a, false, b1, (short)0, acc1, false, false);
        acc2 = __builtin_amdgcn_wmma_f32_16x16x32_bf16(false, a, false, b2, (short)0, acc2, false, false);
        acc3 = __builtin_amdgcn_wmma_f32_16x16x32_bf16(false, a, false, b3, (short)0, acc3, false, false);
    }
    int hi = lane >> 4;
    v8f accs[4] = {acc0, acc1, acc2, acc3};
#pragma unroll
    for (int j = 0; j < 4; ++j) {
        int n = n0 + 16 * j + (lane & 15);
#pragma unroll
        for (int r = 0; r < 8; ++r) {
            int m = m0 + r + 8 * hi;
            Db[(size_t)m * N + n] = (bf16)accs[j][r];
        }
    }
}

// ---------------- K9: out = relu(cat x Wres^T + b), cat built on the fly ------
__global__ __launch_bounds__(256) void gemm_out(const bf16* __restrict__ Ctb,
                                                const bf16* __restrict__ Ab,
                                                const bf16* __restrict__ Bvb,
                                                const bf16* __restrict__ Wb,
                                                const float* __restrict__ brs,
                                                float* __restrict__ out) {
    __shared__ float tile[8][16][17];
    int lane = threadIdx.x & 31;
    int wib = threadIdx.x >> 5;
    int wave = blockIdx.x * 8 + wib;
    const int MT = Lc / 16, NB = H / 64, K = 4 * H;
    int nb = wave % NB, mt = (wave / NB) % MT, b = wave / (NB * MT);
    const bf16* Cb_ = Ctb + (size_t)b * Lc * H;
    const bf16* Ab_ = Ab + (size_t)b * Lc * H;
    const bf16* Bv_ = Bvb + (size_t)b * Lc * H;
    int m0 = mt * 16, n0 = nb * 64;
    int row = m0 + (lane & 15);
    v8f acc0 = {}, acc1 = {}, acc2 = {}, acc3 = {};
#pragma unroll 2
    for (int k0 = 0; k0 < K; k0 += 32) {
        int seg = k0 >> 8;
        int g   = (k0 & 255) + ((lane >> 4) << 3);   // within-segment K offset for A chunks
        size_t ro = (size_t)row * H + g;
        v16bf a;
        if (seg == 0)      a = join8(*(const v8bf*)(Cb_ + ro), *(const v8bf*)(Cb_ + ro + 16));
        else if (seg == 1) a = join8(*(const v8bf*)(Ab_ + ro), *(const v8bf*)(Ab_ + ro + 16));
        else if (seg == 2) a = join8(mul8(Cb_ + ro, Ab_ + ro), mul8(Cb_ + ro + 16, Ab_ + ro + 16));
        else               a = join8(mul8(Cb_ + ro, Bv_ + ro), mul8(Cb_ + ro + 16, Bv_ + ro + 16));
        v16bf b0 = load_b_frag(Wb, K, n0,      k0, lane);
        v16bf b1 = load_b_frag(Wb, K, n0 + 16, k0, lane);
        v16bf b2 = load_b_frag(Wb, K, n0 + 32, k0, lane);
        v16bf b3 = load_b_frag(Wb, K, n0 + 48, k0, lane);
        acc0 = __builtin_amdgcn_wmma_f32_16x16x32_bf16(false, a, false, b0, (short)0, acc0, false, false);
        acc1 = __builtin_amdgcn_wmma_f32_16x16x32_bf16(false, a, false, b1, (short)0, acc1, false, false);
        acc2 = __builtin_amdgcn_wmma_f32_16x16x32_bf16(false, a, false, b2, (short)0, acc2, false, false);
        acc3 = __builtin_amdgcn_wmma_f32_16x16x32_bf16(false, a, false, b3, (short)0, acc3, false, false);
    }
    int hi = lane >> 4;
    int hl = lane >> 1, ch = (lane & 1) * 8;
    v8f accs[4] = {acc0, acc1, acc2, acc3};
#pragma unroll
    for (int j = 0; j < 4; ++j) {
        int h0 = n0 + 16 * j;
        float bias = brs[h0 + (lane & 15)];
#pragma unroll
        for (int r = 0; r < 8; ++r)
            tile[wib][r + 8 * hi][lane & 15] = fmaxf(accs[j][r] + bias, 0.0f);
        // wave-local LDS transpose -> coalesced (B,H,Lc) store
        float v[8];
#pragma unroll
        for (int q = 0; q < 8; ++q) v[q] = tile[wib][ch + q][hl];
        float* dst = out + ((size_t)b * H + h0 + hl) * Lc + m0 + ch;
        *(float4*)(dst)     = make_float4(v[0], v[1], v[2], v[3]);
        *(float4*)(dst + 4) = make_float4(v[4], v[5], v[6], v[7]);
    }
}

extern "C" void kernel_launch(void* const* d_in, const int* in_sizes, int n_in,
                              void* d_out, int out_size, void* d_ws, size_t ws_size,
                              hipStream_t stream) {
    const float* C    = (const float*)d_in[0];
    const float* Q    = (const float*)d_in[1];
    const int*   cmask = (const int*)d_in[2];
    const int*   qmask = (const int*)d_in[3];
    const float* w    = (const float*)d_in[4];
    const float* Wres = (const float*)d_in[5];
    const float* brs  = (const float*)d_in[6];
    float* out = (float*)d_out;

    char* ws = (char*)d_ws;
    size_t off = 0;
    auto alloc = [&](size_t bytes) { off = (off + 255) & ~(size_t)255; size_t o = off; off += bytes; return o; };
    bf16*  Ctb   = (bf16*)(ws + alloc((size_t)B * Lc * H * 2));
    bf16*  CWb   = (bf16*)(ws + alloc((size_t)B * Lc * H * 2));
    bf16*  Cb    = (bf16*)(ws + alloc((size_t)B * H * Lc * 2));
    bf16*  Qtb   = (bf16*)(ws + alloc((size_t)B * Lq * H * 2));
    bf16*  Qb    = (bf16*)(ws + alloc((size_t)B * H * Lq * 2));
    bf16*  Wb    = (bf16*)(ws + alloc((size_t)H * 4 * H * 2));
    float* c1    = (float*)(ws + alloc((size_t)B * Lc * 4));
    float* q2    = (float*)(ws + alloc((size_t)B * Lq * 4));
    float* S     = (float*)(ws + alloc((size_t)B * Lc * Lq * 4));
    bf16*  Srow  = (bf16*)(ws + alloc((size_t)B * Lc * Lq * 2));
    bf16*  ScolT = (bf16*)(ws + alloc((size_t)B * Lq * Lc * 2));
    bf16*  Tt    = (bf16*)(ws + alloc((size_t)B * H * Lq * 2));
    bf16*  Abuf  = (bf16*)(ws + alloc((size_t)B * Lc * H * 2));
    bf16*  Bvb   = (bf16*)(ws + alloc((size_t)B * Lc * H * 2));

    conv_w_kernel<<<(H * 4 * H + 255) / 256, 256, 0, stream>>>(Wres, Wb, H * 4 * H);
    trans_kernel<<<B * (H / 32) * (Lc / 32), 256, 0, stream>>>(C, w + 2 * H, Ctb, CWb, Cb, Lc);
    trans_kernel<<<B * (H / 32) * (Lq / 32), 256, 0, stream>>>(Q, nullptr, Qtb, nullptr, Qb, Lq);
    vec_kernel<<<(B * (Lc + Lq) + 255) / 256, 256, 0, stream>>>(C, Q, w, c1, q2);

    // S = CW x Qt^T + c1 + q2
    gemm_score<<<(B * (Lc / 16) * (Lq / 64)) / 8, 256, 0, stream>>>(CWb, Qtb, c1, q2, S);
    row_softmax<<<(B * Lc) / 8, 256, 0, stream>>>(S, qmask, Srow);
    col_softmax_t<<<B * (Lq / 32), 256, 0, stream>>>(S, cmask, ScolT);

    // T^T(H,Lq) = Cb(H,Lc) x ScolT(Lq,Lc)^T    [= (S_col^T · Ct)^T ]
    gemm_bf<<<(B * (H / 16) * (Lq / 64)) / 8, 256, 0, stream>>>(
        Cb, ScolT, Tt, H, Lq, Lc, (long)H * Lc, (long)Lq * Lc, (long)H * Lq);
    // A(Lc,H) = Srow(Lc,Lq) x Qb(H,Lq)^T
    gemm_bf<<<(B * (Lc / 16) * (H / 64)) / 8, 256, 0, stream>>>(
        Srow, Qb, Abuf, Lc, H, Lq, (long)Lc * Lq, (long)H * Lq, (long)Lc * H);
    // Bv(Lc,H) = Srow(Lc,Lq) x Tt(H,Lq)^T
    gemm_bf<<<(B * (Lc / 16) * (H / 64)) / 8, 256, 0, stream>>>(
        Srow, Tt, Bvb, Lc, H, Lq, (long)Lc * Lq, (long)H * Lq, (long)Lc * H);
    // out = relu(cat x Wres^T + b), transposed store
    gemm_out<<<(B * (Lc / 16) * (H / 64)) / 8, 256, 0, stream>>>(Ctb, Abuf, Bvb, Wb, brs, out);
}
